// DiscreteLatentDistributionVQ_66597762892300
// MI455X (gfx1250) — matrix operand order repaired and checked by reading, not compiled, not run on hardware
//
#include <hip/hip_runtime.h>
#include <hip/hip_bf16.h>
#include <stdint.h>

// Problem constants (from reference)
#define N_ROWS   131072
#define D_IN     512
#define HID      128
#define FEAT     256
#define KCB      512
#define TEMP_INV 10.0f        // 1/0.1
#define ROWS_PER_BLOCK 128    // 8 waves * 16 rows

typedef __attribute__((ext_vector_type(16))) __bf16 v16bf;
typedef __attribute__((ext_vector_type(8)))  __bf16 v8bf;
typedef __attribute__((ext_vector_type(4)))  __bf16 v4bf;
typedef __attribute__((ext_vector_type(8)))  float  v8f;

// ---- LDS layout (bytes, dynamic) -------------------------------------------
constexpr int Z_OFF    = 0;        // 128x256 bf16 z tile (persists)     65536
constexpr int H_OFF    = 65536;    // 128x128 bf16 hidden tile           32768
constexpr int SA_OFF   = 98304;    // 128x32 bf16 stage (X / W2t lo)      8192
constexpr int SB_OFF   = 106496;   // 128x32 bf16 stage (W1t / W2t hi)    8192
constexpr int B1_OFF   = 114688;
constexpr int B2_OFF   = 115200;
constexpr int ZSQ_OFF  = 116224;
constexpr int CSQ_OFF  = 116736;
constexpr int BEST_OFF = 117248;   // 8-byte aligned, 128 x u64
constexpr int IDX_OFF  = 118272;
constexpr int RED_OFF  = 118784;
constexpr int CB0_OFF  = 119808;   // 128x32 f32 codebook stage (async)  16384
constexpr int CB1_OFF  = 136192;   // double buffer                      16384
constexpr int SMEM_BYTES = 152576;

// ---- async global -> LDS copy (CDNA5, ASYNCcnt) ----------------------------
__device__ inline void async_ld16(void* lds_ptr, const void* gptr) {
  // flat-LDS aperture: addr[31:0] is the wave-relative LDS byte address
  unsigned loff = (unsigned)(uintptr_t)lds_ptr;
  asm volatile("global_load_async_to_lds_b128 %0, %1, off"
               :: "v"(loff), "v"(gptr) : "memory");
}
__device__ inline void wait_async_le4() {
#if __has_builtin(__builtin_amdgcn_s_wait_asynccnt)
  __builtin_amdgcn_s_wait_asynccnt(4);
#else
  asm volatile("s_wait_asynccnt 0x4" ::: "memory");
#endif
}
__device__ inline void wait_async_0() {
#if __has_builtin(__builtin_amdgcn_s_wait_asynccnt)
  __builtin_amdgcn_s_wait_asynccnt(0);
#else
  asm volatile("s_wait_asynccnt 0x0" ::: "memory");
#endif
}

// ---- WMMA fragment helpers (contiguous k = 16*half + s for A and B, wave32) --------------------
union V16U { v16bf v; struct { v8bf lo; v8bf hi; } p; };

// 16-bit A matrix 16x32: lanes 0-15 -> M=lane, K=0..15;
// lanes 16-31 -> M=lane-16, K=16..31.
__device__ inline v16bf ldsA32(const __bf16* base, int strideE, int lane, int k0) {
  int m = lane & 15, half = lane >> 4;
  const __bf16* p = base + m * strideE + k0 + half * 16;
  V16U u;
  u.p.lo = *(const v8bf*)(p);
  u.p.hi = *(const v8bf*)(p + 8);
  return u.v;
}

// 16-bit B matrix 32x16 stored transposed in LDS as Bt[n][k]:
// lanes 0-15 -> N=lane, K=0..15; lanes 16-31 -> N=lane-16, K=16..31.
__device__ inline v16bf ldsB32(const __bf16* base, int strideE, int lane) {
  int n = lane & 15, half = lane >> 4;
  const __bf16* p = base + n * strideE + half * 16;
  V16U u;
  u.p.lo = *(const v8bf*)(p);
  u.p.hi = *(const v8bf*)(p + 8);
  return u.v;
}

// B fragment from f32 LDS tile Bt[n][k] (stride in floats); convert to bf16.
__device__ inline v16bf ldsB32_f32(const float* base, int strideF, int lane) {
  int n = lane & 15, half = lane >> 4;
  const float* p = base + n * strideF + half * 16;
  v16bf r;
  #pragma unroll
  for (int i = 0; i < 16; ++i) r[i] = (__bf16)p[i];
  return r;
}

__device__ inline v8f wmma_bf16(v16bf a, v16bf b, v8f c) {
  return __builtin_amdgcn_wmma_f32_16x16x32_bf16(
      false, a, false, b, (short)0, c, false, false);
}

// Deterministic hash-based Gumbel noise (stand-in for threefry(key=42))
__device__ inline float gumbel_noise(unsigned row, unsigned col) {
  unsigned x = row * 0x9E3779B9u + col * 0x85EBCA77u + 0x2545F491u;
  x ^= x >> 16; x *= 0x7FEB352Du; x ^= x >> 15; x *= 0x846CA68Bu; x ^= x >> 16;
  float u = (float)(x >> 8) * (1.0f / 16777216.0f) + 1.0e-12f;
  return -__logf(-__logf(u) + 1.0e-20f);
}

// ---- init: zero ws accumulators (loss sum + K counts) ----------------------
__global__ void vq_init_kernel(float* ws) {
  int t = threadIdx.x + blockIdx.x * blockDim.x;
  if (t < 1 + KCB) ws[t] = 0.0f;
}

// ---- main fused kernel -----------------------------------------------------
__global__ __launch_bounds__(256)
void vq_fused_kernel(const float* __restrict__ X,
                     const float* __restrict__ W1, const float* __restrict__ b1,
                     const float* __restrict__ W2, const float* __restrict__ b2,
                     const float* __restrict__ CB,
                     float* __restrict__ out_q,    // [N,FEAT]
                     float* __restrict__ out_enc,  // [N,K]
                     float* __restrict__ out_nd,   // [N,K]  (-distances)
                     float* __restrict__ ws_loss,  // [1]
                     float* __restrict__ ws_cnt) { // [K]
  extern __shared__ char smem[];
  __bf16* sZ   = (__bf16*)(smem + Z_OFF);
  __bf16* sH   = (__bf16*)(smem + H_OFF);
  __bf16* sA   = (__bf16*)(smem + SA_OFF);
  __bf16* sB   = (__bf16*)(smem + SB_OFF);
  __bf16* sW2t = (__bf16*)(smem + SA_OFF);   // stage2: 256x32 spans sA+sB
  float*  b1s  = (float*)(smem + B1_OFF);
  float*  b2s  = (float*)(smem + B2_OFF);
  float*  zsqs = (float*)(smem + ZSQ_OFF);
  float*  csqs = (float*)(smem + CSQ_OFF);
  unsigned long long* bests = (unsigned long long*)(smem + BEST_OFF);
  int*    idxs = (int*)(smem + IDX_OFF);
  float*  red  = (float*)(smem + RED_OFF);
  float*  sCB0 = (float*)(smem + CB0_OFF);
  float*  sCB1 = (float*)(smem + CB1_OFF);

  const int tid  = threadIdx.x;
  const int lane = tid & 31;
  const int w    = tid >> 5;                  // wave id 0..7 (rows 16w..16w+15)
  const int m    = lane & 15;
  const int half = lane >> 4;
  const size_t rowbase = (size_t)blockIdx.x * ROWS_PER_BLOCK;

  if (tid < HID)  b1s[tid] = b1[tid];
  if (tid < FEAT) b2s[tid] = b2[tid];
  if (tid < ROWS_PER_BLOCK) bests[tid] = 0ull;

  // ===================== Stage 1: h = relu(X @ W1 + b1) =====================
  v8f acc_h[8];
  #pragma unroll
  for (int j = 0; j < 8; ++j) acc_h[j] = (v8f){0,0,0,0,0,0,0,0};

  for (int s = 0; s < D_IN / 32; ++s) {
    __syncthreads();
    // stage X[128 rows][32 k] -> sA (bf16, row-major stride 32)
    for (int i = tid; i < 1024; i += 256) {
      int r = i >> 3, c4 = (i & 7) << 2;
      float4 v = *(const float4*)(X + (rowbase + r) * D_IN + s * 32 + c4);
      *(v4bf*)(sA + r * 32 + c4) =
          (v4bf){(__bf16)v.x, (__bf16)v.y, (__bf16)v.z, (__bf16)v.w};
      if (s + 1 < D_IN / 32)  // warm L2/L0 for next slice (global_prefetch)
        __builtin_prefetch(X + (rowbase + r) * D_IN + (s + 1) * 32 + c4, 0, 1);
    }
    // stage W1t[128 cols][32 k] -> sB (transpose on store)
    for (int i = tid; i < 1024; i += 256) {
      int kr = i >> 5, c4 = (i & 31) << 2;
      float4 v = *(const float4*)(W1 + (size_t)(s * 32 + kr) * HID + c4);
      sB[(c4 + 0) * 32 + kr] = (__bf16)v.x;
      sB[(c4 + 1) * 32 + kr] = (__bf16)v.y;
      sB[(c4 + 2) * 32 + kr] = (__bf16)v.z;
      sB[(c4 + 3) * 32 + kr] = (__bf16)v.w;
    }
    __syncthreads();
    v16bf a = ldsA32(sA + (w * 16) * 32, 32, lane, 0);
    #pragma unroll
    for (int g = 0; g < 2; ++g) {       // batch frag loads, then WMMAs
      v16bf bf[4];
      #pragma unroll
      for (int jj = 0; jj < 4; ++jj)
        bf[jj] = ldsB32(sB + ((g * 4 + jj) * 16) * 32, 32, lane);
      #pragma unroll
      for (int jj = 0; jj < 4; ++jj)
        acc_h[g * 4 + jj] = wmma_bf16(a, bf[jj], acc_h[g * 4 + jj]);
    }
  }
  // write h = relu(acc + b1) to LDS (C layout: M = r + 8*half, N = m)
  #pragma unroll
  for (int j = 0; j < 8; ++j) {
    int col = j * 16 + m;
    #pragma unroll
    for (int r = 0; r < 8; ++r) {
      int row = w * 16 + r + 8 * half;
      float v = acc_h[j][r] + b1s[col];
      sH[row * HID + col] = (__bf16)fmaxf(v, 0.0f);
    }
  }

  // ===================== Stage 2: z = h @ W2 + b2 ===========================
  v8f acc_z[16];
  #pragma unroll
  for (int j = 0; j < 16; ++j) acc_z[j] = (v8f){0,0,0,0,0,0,0,0};

  for (int s = 0; s < HID / 32; ++s) {
    __syncthreads();
    // stage W2t[256 cols][32 k] -> sW2t (16 KB, transpose on store)
    for (int i = tid; i < 2048; i += 256) {
      int kr = i >> 6, c4 = (i & 63) << 2;
      float4 v = *(const float4*)(W2 + (size_t)(s * 32 + kr) * FEAT + c4);
      sW2t[(c4 + 0) * 32 + kr] = (__bf16)v.x;
      sW2t[(c4 + 1) * 32 + kr] = (__bf16)v.y;
      sW2t[(c4 + 2) * 32 + kr] = (__bf16)v.z;
      sW2t[(c4 + 3) * 32 + kr] = (__bf16)v.w;
    }
    __syncthreads();
    v16bf a = ldsA32(sH + (w * 16) * HID, HID, lane, s * 32);
    #pragma unroll
    for (int g = 0; g < 4; ++g) {
      v16bf bf[4];
      #pragma unroll
      for (int jj = 0; jj < 4; ++jj)
        bf[jj] = ldsB32(sW2t + ((g * 4 + jj) * 16) * 32, 32, lane);
      #pragma unroll
      for (int jj = 0; jj < 4; ++jj)
        acc_z[g * 4 + jj] = wmma_bf16(a, bf[jj], acc_z[g * 4 + jj]);
    }
  }
  __syncthreads();
  #pragma unroll
  for (int j = 0; j < 16; ++j) {
    int col = j * 16 + m;
    #pragma unroll
    for (int r = 0; r < 8; ++r) {
      int row = w * 16 + r + 8 * half;
      sZ[row * FEAT + col] = (__bf16)(acc_z[j][r] + b2s[col]);
    }
  }
  __syncthreads();

  // per-row ||z||^2
  if (tid < ROWS_PER_BLOCK) {
    float s = 0.0f;
    for (int f = 0; f < FEAT; ++f) {
      float x = (float)sZ[tid * FEAT + f];
      s += x * x;
    }
    zsqs[tid] = s;
  }

  // ======= Distances + Gumbel-argmax over K (4 chunks of 128 codes) ========
  // Codebook slices stream via async global->LDS (f32, raw), double-buffered:
  // issue slice s+1 while computing slice s; drain with s_wait_asynccnt 4.
  {
    // prologue: issue chunk 0, step 0 into sCB0 (4 x b128 per thread)
    #pragma unroll
    for (int r = 0; r < 4; ++r) {
      int i = tid + (r << 8);
      int colr = i >> 3, f4 = (i & 7) << 2;
      async_ld16(sCB0 + colr * 32 + f4, CB + (size_t)colr * FEAT + f4);
    }
  }

  for (int c = 0; c < KCB / 128; ++c) {
    __syncthreads();
    if (tid < 128) csqs[tid] = 0.0f;
    v8f accd[8];
    #pragma unroll
    for (int j = 0; j < 8; ++j) accd[j] = (v8f){0,0,0,0,0,0,0,0};

    for (int s = 0; s < FEAT / 32; ++s) {
      float* cur = (s & 1) ? sCB1 : sCB0;
      float* nxt = (s & 1) ? sCB0 : sCB1;
      __syncthreads();  // all readers of `nxt` (prev step) are done
      bool hasNext = (s < 7) || (c < (KCB / 128 - 1));
      if (hasNext) {
        int cn = (s < 7) ? c : c + 1;
        int sn = (s < 7) ? s + 1 : 0;
        #pragma unroll
        for (int r = 0; r < 4; ++r) {
          int i = tid + (r << 8);
          int colr = i >> 3, f4 = (i & 7) << 2;
          async_ld16(nxt + colr * 32 + f4,
                     CB + (size_t)(cn * 128 + colr) * FEAT + sn * 32 + f4);
        }
        wait_async_le4();   // newest 4 (nxt) may remain in flight; cur complete
      } else {
        wait_async_0();
      }
      __syncthreads();      // cur visible to all waves

      if (tid < 128) {      // accumulate ||c||^2 from staged f32
        float sv = 0.0f;
        #pragma unroll
        for (int f = 0; f < 32; ++f) {
          float x = cur[tid * 32 + f];
          sv += x * x;
        }
        csqs[tid] += sv;
      }
      v16bf a = ldsA32(sZ + (w * 16) * FEAT, FEAT, lane, s * 32);
      #pragma unroll
      for (int g = 0; g < 2; ++g) {
        v16bf bf[4];
        #pragma unroll
        for (int jj = 0; jj < 4; ++jj)
          bf[jj] = ldsB32_f32(cur + ((g * 4 + jj) * 16) * 32, 32, lane);
        #pragma unroll
        for (int jj = 0; jj < 4; ++jj)
          accd[g * 4 + jj] = wmma_bf16(a, bf[jj], accd[g * 4 + jj]);
      }
    }
    __syncthreads();  // csqs complete

    // -dist = 2*dot - ||z||^2 - ||c||^2 ; Gumbel-max sample via ds_max_u64
    #pragma unroll
    for (int r = 0; r < 8; ++r) {
      int rowL = w * 16 + r + 8 * half;
      size_t rowG = rowbase + rowL;
      float zq = zsqs[rowL];
      float bestv = -3.0e38f;
      int bestc = c * 128;
      #pragma unroll
      for (int j = 0; j < 8; ++j) {
        int colL = c * 128 + j * 16 + m;
        float nd = 2.0f * accd[j][r] - zq - csqs[j * 16 + m];
        out_nd[rowG * KCB + colL] = nd;
        float sc = nd * TEMP_INV + gumbel_noise((unsigned)rowG, (unsigned)colL);
        if (sc > bestv) { bestv = sc; bestc = colL; }
      }
      unsigned ub = __float_as_uint(bestv);
      ub = (ub & 0x80000000u) ? ~ub : (ub | 0x80000000u);  // order-preserving
      unsigned long long pk = ((unsigned long long)ub << 32) | (unsigned)bestc;
      atomicMax(&bests[rowL], pk);
    }
  }
  __syncthreads();

  // extract sampled indices + global histogram
  if (tid < ROWS_PER_BLOCK) {
    int id = (int)(bests[tid] & 0xFFFFFFFFull);
    idxs[tid] = id;
    atomicAdd(&ws_cnt[id], 1.0f);
  }
  __syncthreads();

  // one-hot encodings (zero-fill + 1.0 at idx), float4 stores
  {
    float* enc = out_enc + rowbase * KCB;
    for (int i = tid; i < ROWS_PER_BLOCK * (KCB / 4); i += 256) {
      int r = i >> 7, c4 = (i & 127) << 2;
      int id = idxs[r];
      float4 v = make_float4(0.f, 0.f, 0.f, 0.f);
      if (id >= c4 && id < c4 + 4) (&v.x)[id - c4] = 1.0f;
      *(float4*)(enc + (size_t)r * KCB + c4) = v;
    }
  }

  // quantized_st forward value = codebook[idx]; loss partial = sum (q - z)^2
  float lsum = 0.0f;
  for (int r = 0; r < ROWS_PER_BLOCK; ++r) {
    int id = idxs[r];
    float q = CB[(size_t)id * FEAT + tid];
    float zv = (float)sZ[r * FEAT + tid];
    out_q[(rowbase + r) * FEAT + tid] = q;
    float d = q - zv;
    lsum += d * d;
  }
  red[tid] = lsum;
  __syncthreads();
  for (int o = 128; o > 0; o >>= 1) {
    if (tid < o) red[tid] += red[tid + o];
    __syncthreads();
  }
  if (tid == 0) atomicAdd(ws_loss, red[0]);
}

// ---- finalize: loss scale + perplexity -------------------------------------
__global__ void vq_finalize_kernel(const float* __restrict__ ws_loss,
                                   const float* __restrict__ ws_cnt,
                                   float* __restrict__ out) {
  __shared__ float red[256];
  int t = threadIdx.x;
  float s = 0.0f;
  for (int k = t; k < KCB; k += 256) {
    float p = ws_cnt[k] * (1.0f / (float)N_ROWS);
    s += p * __logf(p + 1e-10f);
  }
  red[t] = s;
  __syncthreads();
  for (int o = 128; o > 0; o >>= 1) {
    if (t < o) red[t] += red[t + o];
    __syncthreads();
  }
  if (t == 0) {
    // q_latent_loss and e_latent_loss are value-identical -> 2 * mean((q-z)^2)
    out[0] = 2.0f * ws_loss[0] / ((float)N_ROWS * (float)FEAT);
    out[(size_t)1 + (size_t)N_ROWS * FEAT] = __expf(-red[0]);  // perplexity
  }
}

// ---- launcher --------------------------------------------------------------
extern "C" void kernel_launch(void* const* d_in, const int* in_sizes, int n_in,
                              void* d_out, int out_size, void* d_ws, size_t ws_size,
                              hipStream_t stream) {
  const float* X  = (const float*)d_in[0];
  const float* W1 = (const float*)d_in[1];
  const float* b1 = (const float*)d_in[2];
  const float* W2 = (const float*)d_in[3];
  const float* b2 = (const float*)d_in[4];
  const float* CB = (const float*)d_in[5];

  float* out = (float*)d_out;
  // layout: [loss(1)][quantized_st(N*FEAT)][perplexity(1)][encodings(N*K)][-distances(N*K)]
  float* out_q   = out + 1;
  float* out_enc = out + 2 + (size_t)N_ROWS * FEAT;
  float* out_nd  = out_enc + (size_t)N_ROWS * KCB;

  float* ws      = (float*)d_ws;   // [0]=loss sum, [1..512]=counts
  float* ws_loss = ws;
  float* ws_cnt  = ws + 1;

  (void)hipFuncSetAttribute((const void*)vq_fused_kernel,
                            hipFuncAttributeMaxDynamicSharedMemorySize,
                            SMEM_BYTES);

  vq_init_kernel<<<3, 256, 0, stream>>>(ws);
  vq_fused_kernel<<<N_ROWS / ROWS_PER_BLOCK, 256, SMEM_BYTES, stream>>>(
      X, W1, b1, W2, b2, CB, out_q, out_enc, out_nd, ws_loss, ws_cnt);
  vq_finalize_kernel<<<1, 256, 0, stream>>>(ws_loss, ws_cnt, out);
}
